// GAT_E_to_R_78950088835240
// MI455X (gfx1250) — matrix operand order repaired and checked by reading, not compiled, not run on hardware
//
#include <hip/hip_runtime.h>

#define R_CONST 500
#define HID 128
#define EPB 64                 // edges per block in edge_gemm (4 tiles of 16)
#define TROW (HID + 8)         // padded bf16 per edge-tile row (bank rotation)
#define MROW (256 + 8)         // padded bf16 per mean-tile row

// padded-LDS dword offsets of the weight pack staged by TDM
// raw: WhT [128x128] | WtT [128x128] | Wsr1T [128x256]  (bf16, contiguous, 128KB)
// TDM pad: +4 dwords every 64 dwords  -> padded(D) = D + 4*(D>>6)
#define LDSW_DWORDS 34816      // 32768 + 4*(32768/64)
#define WT_OFF_D    8704       // 8192  + 4*128
#define WSR1_OFF_D  17408      // 16384 + 4*256

typedef __attribute__((ext_vector_type(16))) __bf16 v16bf;
typedef __attribute__((ext_vector_type(8)))  float  v8f;
typedef __attribute__((ext_vector_type(4)))  unsigned u32x4;
typedef __attribute__((ext_vector_type(8)))  unsigned u32x8;

union FragAB { v16bf v; unsigned u[8]; __bf16 h[16]; };
union FragC  { v8f v; float f[8]; };

// ---- monotonic float<->uint mapping so atomicMax works on signed floats ----
__device__ __forceinline__ unsigned enc_f32(float f) {
  unsigned u = __float_as_uint(f);
  return (u & 0x80000000u) ? ~u : (u | 0x80000000u);
}
__device__ __forceinline__ float dec_f32(unsigned k) {
  unsigned u = (k & 0x80000000u) ? (k & 0x7FFFFFFFu) : ~k;
  return __uint_as_float(u);
}

// ---- WMMA (CDNA5 bf16 16x16x32, fp32 accumulate) ----
__device__ __forceinline__ v8f wmma_bf16(v16bf a, v16bf b, v8f c) {
  return __builtin_amdgcn_wmma_f32_16x16x32_bf16(false, a, false, b, (short)0, c,
                                                 false, false);
}

// A fragment from an LDS row (row = lane&15), packed bf16 pairs per dword.
// Lanes 0-15: u0..7 = K 0..15 ; lanes 16-31: u0..7 = K 16..31.
__device__ __forceinline__ v16bf load_a_pairs(const unsigned* rowp, int k0, int hi) {
  FragAB f;
  int o = (k0 >> 1) + 8 * hi;
  f.u[0] = rowp[o + 0]; f.u[1] = rowp[o + 1]; f.u[2] = rowp[o + 2]; f.u[3] = rowp[o + 3];
  f.u[4] = rowp[o + 4]; f.u[5] = rowp[o + 5]; f.u[6] = rowp[o + 6]; f.u[7] = rowp[o + 7];
  return f.v;
}

// B fragment from the TDM-staged, padded LDS weight pack.
// Matrix stored transposed [N][K] bf16; lane n = column; lanes 0-15 K 0..15,
// lanes 16-31 K 16..31 (pairs per dword). D = raw dword offset in matrix.
__device__ __forceinline__ v16bf load_b_lds(const unsigned* ldsw, int baseD,
                                            int K, int col, int k) {
  int D = col * (K >> 1) + (k >> 1);
  const unsigned* p = ldsw + baseD + D + ((D >> 6) << 2);  // apply TDM padding
  const uint4* q = (const uint4*)p;                        // 16B aligned
  uint4 q0 = q[0], q1 = q[1];
  FragAB f;
  f.u[0] = q0.x; f.u[1] = q0.y; f.u[2] = q0.z; f.u[3] = q0.w;
  f.u[4] = q1.x; f.u[5] = q1.y; f.u[6] = q1.z; f.u[7] = q1.w;
  return f.v;
}

// B fragment straight from a global transposed bf16 weight [N][K] (small GEMM path)
__device__ __forceinline__ v16bf load_b_glb(const __bf16* WT, int K, int col, int k) {
  const unsigned* p = (const unsigned*)(WT + (size_t)col * K + k);
  FragAB f;
#pragma unroll
  for (int i = 0; i < 8; ++i) f.u[i] = p[i];
  return f.v;
}

// ================= kernels =================

__global__ void init_ws(float* cnt, float* sumh, float* sumt,
                        unsigned* segmax, float* segsum, float* outacc) {
  int i = blockIdx.x * 256 + threadIdx.x;
  if (i < R_CONST) cnt[i] = 0.f;
  if (i < R_CONST * HID) { sumh[i] = 0.f; sumt[i] = 0.f; outacc[i] = 0.f; }
  if (i < 3 * R_CONST) { segmax[i] = 0u; segsum[i] = 0.f; }
}

// transpose + convert weights to bf16 [N][K]
__global__ void wconv(const float* __restrict__ Wsr, const float* __restrict__ Wsr1,
                      const float* __restrict__ Wh, const float* __restrict__ Wt,
                      __bf16* __restrict__ WsrT, __bf16* __restrict__ Wsr1T,
                      __bf16* __restrict__ WhT, __bf16* __restrict__ WtT) {
  int i = blockIdx.x * 256 + threadIdx.x;
  if (i < 256 * HID) {
    int k = i / HID, n = i % HID;
    WsrT[n * 256 + k]  = (__bf16)Wsr[i];
    Wsr1T[n * 256 + k] = (__bf16)Wsr1[i];
  }
  if (i < HID * HID) {
    int k = i / HID, n = i % HID;
    WhT[n * HID + k] = (__bf16)Wh[i];
    WtT[n * HID + k] = (__bf16)Wt[i];
  }
}

// per-relation sums of head/tail features + counts
__global__ __launch_bounds__(128)
void seg_sums(const float* __restrict__ x_e, const int* __restrict__ eidx,
              const int* __restrict__ rel, float* __restrict__ cnt,
              float* __restrict__ sumh, float* __restrict__ sumt, int E) {
  int t = threadIdx.x;
  int base = blockIdx.x * 8;
#pragma unroll 1
  for (int j = 0; j < 8; ++j) {
    int e = base + j;
    if (e >= E) break;
    int r = rel[e];
    int h = eidx[e], tt = eidx[E + e];
    atomicAdd(&sumh[r * HID + t], x_e[(size_t)h * HID + t]);
    atomicAdd(&sumt[r * HID + t], x_e[(size_t)tt * HID + t]);
    if (t == 0) atomicAdd(&cnt[r], 1.0f);
  }
}

// y = [mean_h | mean_t] @ W_sr + b_sr  (R x 256 x 128, WMMA, LDS-staged A)
__global__ __launch_bounds__(256)
void rel_gemm(const float* __restrict__ sumh, const float* __restrict__ sumt,
              const float* __restrict__ cnt, const float* __restrict__ b_sr,
              const __bf16* __restrict__ WsrT, float* __restrict__ y) {
  __shared__ __align__(16) __bf16 sm[16][MROW];
  int tid = threadIdx.x;
  // stage 16x256 mean tile as bf16 (branch-free)
  {
    int m = tid >> 4;
    int c0 = (tid & 15) * 16;
    int row = blockIdx.x * 16 + m;
    bool vrow = row < R_CONST;
    float rc = vrow ? 1.0f / fmaxf(cnt[vrow ? row : 0], 1.0f) : 0.f;
    const float* ph = sumh + (size_t)(vrow ? row : 0) * HID;
    const float* pt = sumt + (size_t)(vrow ? row : 0) * HID;
#pragma unroll
    for (int j = 0; j < 16; ++j) {
      int k = c0 + j;
      float v = (k < HID ? ph[k] : pt[k - HID]) * rc;
      sm[m][k] = (__bf16)(vrow ? v : 0.f);
    }
  }
  __syncthreads();

  int lane = tid & 31, wave = tid >> 5, hi = lane >> 4;
  int col = wave * 16 + (lane & 15);
  const unsigned* rowp = (const unsigned*)&sm[lane & 15][0];
  v8f acc = {0.f, 0.f, 0.f, 0.f, 0.f, 0.f, 0.f, 0.f};
#pragma unroll
  for (int k0 = 0; k0 < 256; k0 += 32)
    acc = wmma_bf16(load_a_pairs(rowp, k0, hi),
                    load_b_glb(WsrT, 256, col, k0 + 16 * hi), acc);
  float bias = b_sr[col];
  FragC c; c.v = acc;
#pragma unroll
  for (int v = 0; v < 8; ++v) {
    int row = blockIdx.x * 16 + v + 8 * hi;
    if (row < R_CONST) y[row * HID + col] = c.f[v] + bias;
  }
}

// z[r] = y[r] . a_t3
__global__ __launch_bounds__(128)
void rel_z(const float* __restrict__ y, const float* __restrict__ a_t3,
           float* __restrict__ z) {
  __shared__ float red[128];
  int r = blockIdx.x, t = threadIdx.x;
  red[t] = y[r * HID + t] * a_t3[t];
  __syncthreads();
  for (int s = 64; s > 0; s >>= 1) {
    if (t < s) red[t] += red[t + s];
    __syncthreads();
  }
  if (t == 0) z[r] = red[0];
}

// fused edge GEMMs + attention scores + segment max.
// Weight pack staged into LDS once per block via TDM (tensor_load_to_lds).
__global__ __launch_bounds__(256)
void edge_gemm(const float* __restrict__ x_e, const int* __restrict__ eidx,
               const int* __restrict__ rel, const float* __restrict__ b_sr1,
               const float* __restrict__ a_h1, const float* __restrict__ a_h2,
               const float* __restrict__ a_t1, const float* __restrict__ a_t2,
               const float* __restrict__ a_h3, const float* __restrict__ zrel,
               const __bf16* __restrict__ Wpack,
               float* __restrict__ xrh, float* __restrict__ xrt, float* __restrict__ xr2,
               float* __restrict__ e1, float* __restrict__ e2, float* __restrict__ e3,
               unsigned* __restrict__ segmax, int E) {
  __shared__ __align__(16) unsigned lds_w[LDSW_DWORDS];   // 136KB weight pack (padded)
  __shared__ __align__(16) __bf16 sh_h[16][TROW];
  __shared__ __align__(16) __bf16 sh_t[16][TROW];
  __shared__ float sE[3][16];
  __shared__ int s_rel[16];
  int tid = threadIdx.x;

  // ---- wave 0 issues one TDM load of the 128KB bf16 weight pack ----
  if (tid < 32) {
    unsigned long long ga = (unsigned long long)(size_t)Wpack;
    unsigned lds_off = (unsigned)(size_t)(void*)&lds_w[0];  // low 32 bits = LDS offset
    u32x4 g0;
    g0[0] = 1u;                                            // count=1 valid descriptor
    g0[1] = __builtin_amdgcn_readfirstlane(lds_off);       // lds_addr
    g0[2] = __builtin_amdgcn_readfirstlane((unsigned)ga);  // global_addr[31:0]
    g0[3] = __builtin_amdgcn_readfirstlane(
              ((unsigned)(ga >> 32) & 0x01FFFFFFu) | (2u << 30)); // addr[56:32]|type=2
    u32x8 g1;
    // data_size=8B, pad_enable, pad_interval=64 dwords (code 5), pad_amount=4 dwords (code 3)
    g1[0] = (3u << 16) | (1u << 20) | (5u << 22) | (3u << 25);
    g1[1] = (16384u & 0xFFFFu) << 16;        // tensor_dim0 = 16384 (8B units), low 16
    g1[2] = (16384u >> 16) | (1u << 16);     // tensor_dim0 hi | tensor_dim1 = 1
    g1[3] = (16384u & 0xFFFFu) << 16;        // tile_dim0 = 16384
    g1[4] = 1u;                              // tile_dim1 = 1, tile_dim2 = 0
    g1[5] = 16384u;                          // tensor_dim0_stride low 32
    g1[6] = 0u;
    g1[7] = 0u;
    asm volatile("tensor_load_to_lds %0, %1" :: "s"(g0), "s"(g1) : "memory");
  }

  int lane = tid & 31, wave = tid >> 5, hi = lane >> 4;
  int col = wave * 16 + (lane & 15);
  float bias = b_sr1[col];
  float c_h1 = a_h1[col], c_h2 = a_h2[col];
  float c_t1 = a_t1[col], c_t2 = a_t2[col], c_h3 = a_h3[col];
  const unsigned* rowh = (const unsigned*)&sh_h[lane & 15][0];
  const unsigned* rowt = (const unsigned*)&sh_t[lane & 15][0];

#pragma unroll 1
  for (int it = 0; it < EPB / 16; ++it) {
    int ebase = blockIdx.x * EPB + it * 16;
    if (tid < 48) sE[tid >> 4][tid & 15] = 0.f;

    // gather 16 edges' head/tail rows into LDS as bf16
    {
      int m = tid >> 4;
      int c0 = (tid & 15) * 8;
      int edge = ebase + m;
      if (edge < E) {
        int h = eidx[edge], t = eidx[E + edge];
        const float* ph = x_e + (size_t)h * HID + c0;
        const float* pt = x_e + (size_t)t * HID + c0;
#pragma unroll
        for (int j = 0; j < 8; ++j) {
          sh_h[m][c0 + j] = (__bf16)ph[j];
          sh_t[m][c0 + j] = (__bf16)pt[j];
        }
        if ((tid & 15) == 0) s_rel[m] = rel[edge];
      } else {
#pragma unroll
        for (int j = 0; j < 8; ++j) {
          sh_h[m][c0 + j] = (__bf16)0.f;
          sh_t[m][c0 + j] = (__bf16)0.f;
        }
      }
    }
    if (it == 0 && tid < 32) __builtin_amdgcn_s_wait_tensorcnt(0);  // TDM done
    __syncthreads();

    v8f acc_h = {0.f, 0.f, 0.f, 0.f, 0.f, 0.f, 0.f, 0.f};
    v8f acc_t = acc_h, acc_r2 = acc_h;
#pragma unroll
    for (int k0 = 0; k0 < HID; k0 += 32) {
      v16bf ah = load_a_pairs(rowh, k0, hi);
      v16bf at = load_a_pairs(rowt, k0, hi);
      int kk = k0 + 16 * hi;
      acc_h  = wmma_bf16(ah, load_b_lds(lds_w, 0,          HID, col, kk), acc_h);
      acc_t  = wmma_bf16(at, load_b_lds(lds_w, WT_OFF_D,   HID, col, kk), acc_t);
      acc_r2 = wmma_bf16(at, load_b_lds(lds_w, WSR1_OFF_D, 256, col, kk), acc_r2);
      acc_r2 = wmma_bf16(ah, load_b_lds(lds_w, WSR1_OFF_D, 256, col, HID + kk), acc_r2);
    }

    FragC ch, ct, cr;
    ch.v = acc_h; ct.v = acc_t; cr.v = acc_r2;
#pragma unroll
    for (int v = 0; v < 8; ++v) {
      float fh = ch.f[v], ft = ct.f[v], fr = cr.f[v] + bias;
      float sv = fh + fr + ft;
      int mm = v + 8 * hi;
      atomicAdd(&sE[0][mm], fh * c_h1 + sv * c_h2);
      atomicAdd(&sE[1][mm], sv * c_t1 + ft * c_t2);
      atomicAdd(&sE[2][mm], sv * c_h3);
      int eg = ebase + mm;
      if (eg < E) {
        size_t o = (size_t)eg * HID + col;
        xrh[o] = fh; xrt[o] = ft; xr2[o] = fr;
      }
    }
    __syncthreads();

    if (tid < 16) {
      int eg = ebase + tid;
      if (eg < E) {
        int r = s_rel[tid];
        float f1 = sE[0][tid], f2 = sE[1][tid], f3 = sE[2][tid] + zrel[r];
        f1 = f1 > 0.f ? f1 : 0.01f * f1;
        f2 = f2 > 0.f ? f2 : 0.01f * f2;
        f3 = f3 > 0.f ? f3 : 0.01f * f3;
        e1[eg] = f1; e2[eg] = f2; e3[eg] = f3;
        atomicMax(&segmax[0 * R_CONST + r], enc_f32(f1));
        atomicMax(&segmax[1 * R_CONST + r], enc_f32(f2));
        atomicMax(&segmax[2 * R_CONST + r], enc_f32(f3));
      }
    }
    __syncthreads();
  }
}

// exp(e - segmax) and segment denominators
__global__ void seg_exp(const int* __restrict__ rel, float* __restrict__ e1,
                        float* __restrict__ e2, float* __restrict__ e3,
                        const unsigned* __restrict__ segmax,
                        float* __restrict__ segsum, int E) {
  int e = blockIdx.x * 256 + threadIdx.x;
  if (e >= E) return;
  int r = rel[e];
  float x1 = __expf(e1[e] - dec_f32(segmax[r]));
  float x2 = __expf(e2[e] - dec_f32(segmax[R_CONST + r]));
  float x3 = __expf(e3[e] - dec_f32(segmax[2 * R_CONST + r]));
  e1[e] = x1; e2[e] = x2; e3[e] = x3;
  atomicAdd(&segsum[r], x1);
  atomicAdd(&segsum[R_CONST + r], x2);
  atomicAdd(&segsum[2 * R_CONST + r], x3);
}

// out_acc[r] += alpha1*x_r_h + alpha2*x_r_t   (alpha3 spmm == y[r], folded later)
__global__ __launch_bounds__(128)
void spmm(const int* __restrict__ rel, const float* __restrict__ e1,
          const float* __restrict__ e2, const float* __restrict__ segsum,
          const float* __restrict__ xrh, const float* __restrict__ xrt,
          float* __restrict__ outacc, int E) {
  int t = threadIdx.x;
  int base = blockIdx.x * 8;
#pragma unroll 1
  for (int j = 0; j < 8; ++j) {
    int e = base + j;
    if (e >= E) break;
    int r = rel[e];
    float a1 = e1[e] / segsum[r];
    float a2 = e2[e] / segsum[R_CONST + r];
    float val = a1 * xrh[(size_t)e * HID + t] + a2 * xrt[(size_t)e * HID + t];
    atomicAdd(&outacc[r * HID + t], val);
  }
}

// out = (out_acc + y)[rel] + y[rel] + x_res2
__global__ void finalize(const int* __restrict__ rel, const float* __restrict__ outacc,
                         const float* __restrict__ y, const float* __restrict__ xr2,
                         float* __restrict__ out, int E) {
  int i = blockIdx.x * 256 + threadIdx.x;
  if (i >= E * HID) return;
  int e = i >> 7, f = i & (HID - 1);
  int r = rel[e];
  float yv = y[r * HID + f];
  out[i] = outacc[r * HID + f] + yv + yv + xr2[i];
}

// ================= launcher =================

extern "C" void kernel_launch(void* const* d_in, const int* in_sizes, int n_in,
                              void* d_out, int out_size, void* d_ws, size_t ws_size,
                              hipStream_t stream) {
  const float* x_e   = (const float*)d_in[0];
  const int*   eidx  = (const int*)d_in[1];
  const int*   rel   = (const int*)d_in[2];
  const float* W_sr  = (const float*)d_in[4];
  const float* b_sr  = (const float*)d_in[5];
  const float* W_sr1 = (const float*)d_in[6];
  const float* b_sr1 = (const float*)d_in[7];
  const float* W_h   = (const float*)d_in[8];
  const float* W_t   = (const float*)d_in[9];
  const float* a_h1  = (const float*)d_in[10];
  const float* a_h2  = (const float*)d_in[11];
  const float* a_t1  = (const float*)d_in[12];
  const float* a_t2  = (const float*)d_in[13];
  const float* a_h3  = (const float*)d_in[14];
  const float* a_t3  = (const float*)d_in[15];
  float* out = (float*)d_out;

  const int E = in_sizes[2];
  const int R = R_CONST;

  char* w = (char*)d_ws;
  auto carve = [&](size_t bytes) {
    void* p = (void*)w;
    w += (bytes + 255) & ~(size_t)255;
    return p;
  };
  float*    cnt    = (float*)carve((size_t)R * 4);
  float*    sumh   = (float*)carve((size_t)R * HID * 4);
  float*    sumt   = (float*)carve((size_t)R * HID * 4);
  float*    y      = (float*)carve((size_t)R * HID * 4);
  float*    z      = (float*)carve((size_t)R * 4);
  unsigned* segmax = (unsigned*)carve((size_t)3 * R * 4);
  float*    segsum = (float*)carve((size_t)3 * R * 4);
  float*    outacc = (float*)carve((size_t)R * HID * 4);
  __bf16*   WsrT   = (__bf16*)carve((size_t)256 * HID * 2);
  // contiguous 128KB pack for TDM: WhT | WtT | Wsr1T (each size multiple of 256B)
  __bf16*   Wpack  = (__bf16*)carve((size_t)(HID * HID * 2 + HID * HID * 2 + 256 * HID * 2));
  __bf16*   WhT    = Wpack;
  __bf16*   WtT    = Wpack + HID * HID;
  __bf16*   Wsr1T  = Wpack + 2 * HID * HID;
  float*    xrh    = (float*)carve((size_t)E * HID * 4);
  float*    xrt    = (float*)carve((size_t)E * HID * 4);
  float*    xr2    = (float*)carve((size_t)E * HID * 4);
  float*    e1     = (float*)carve((size_t)E * 4);
  float*    e2     = (float*)carve((size_t)E * 4);
  float*    e3     = (float*)carve((size_t)E * 4);
  (void)ws_size; (void)n_in; (void)out_size;

  init_ws<<<(R * HID + 255) / 256, 256, 0, stream>>>(cnt, sumh, sumt, segmax, segsum, outacc);
  wconv<<<(256 * HID + 255) / 256, 256, 0, stream>>>(W_sr, W_sr1, W_h, W_t,
                                                     WsrT, Wsr1T, WhT, WtT);
  seg_sums<<<(E + 7) / 8, 128, 0, stream>>>(x_e, eidx, rel, cnt, sumh, sumt, E);
  rel_gemm<<<(R + 15) / 16, 256, 0, stream>>>(sumh, sumt, cnt, b_sr, WsrT, y);
  rel_z<<<R, 128, 0, stream>>>(y, a_t3, z);
  edge_gemm<<<(E + EPB - 1) / EPB, 256, 0, stream>>>(x_e, eidx, rel, b_sr1,
                                                     a_h1, a_h2, a_t1, a_t2, a_h3, z,
                                                     Wpack,
                                                     xrh, xrt, xr2, e1, e2, e3, segmax, E);
  seg_exp<<<(E + 255) / 256, 256, 0, stream>>>(rel, e1, e2, e3, segmax, segsum, E);
  spmm<<<(E + 7) / 8, 128, 0, stream>>>(rel, e1, e2, segsum, xrh, xrt, outacc, E);
  finalize<<<((size_t)E * HID + 255) / 256, 256, 0, stream>>>(rel, outacc, y, xr2, out, E);
}
